// TrainableMPOLayer_65420941853375
// MI455X (gfx1250) — compile-verified, matrix operand order checked by reading, not run on hardware
//
#include <hip/hip_runtime.h>

typedef __attribute__((ext_vector_type(16))) _Float16 v16h;
typedef __attribute__((ext_vector_type(8)))  _Float16 v8h;
typedef __attribute__((ext_vector_type(2)))  _Float16 h2;
typedef __attribute__((ext_vector_type(8)))  float    v8f;

#define KP   40     // padded K stride (halfs) for K=32 tiles
#define KP2  520    // padded K stride (halfs) for K=512 rows

#define C1BUF_HALFS (16 * KP2)      // 8320 halfs per core1 tile buffer
#define C1BUF_BYTES (C1BUF_HALFS * 2)

// LDS layout (offsets in halfs); total 145920 halfs = 291840 bytes (< 320KB)
#define OFF_XA   0          // XA   [256 rows=(j2*16+j3)][KP]   A for stage 1
#define OFF_B0   10240      // B0L  [8 i1][32 r1][KP]           B for stage 1
#define OFF_B2   20480      // B2L  [32 r2][16 i3][KP]          B for stage 3
#define OFF_T1   40960      // T1A  [8 i1][16 m=j3][KP2]        A for stage 2
#define OFF_C1   107520     // C1B  [4 bufs][16 n=i2][KP2]      quad-buffered B tiles
#define OFF_SC   140800     // SCR  [8 wave][16 m=i2][KP]       A scratch stage 3
#define LDS_HALFS 145920
#define LDS_BYTES (LDS_HALFS * 2)

static __device__ __forceinline__ v16h cat16(v8h lo, v8h hi) {
  return __builtin_shufflevector(lo, hi, 0,1,2,3,4,5,6,7,8,9,10,11,12,13,14,15);
}

// wave32 16-bit A/B fragment: per lane, 8 halfs at p and 8 at p+16
// (ISA 16x32 layout: half h covers K = h*8..h*8+7 and 16+h*8..+7)
static __device__ __forceinline__ v16h frag(const _Float16* p) {
  v8h lo = *(const v8h*)p;
  v8h hi = *(const v8h*)(p + 16);
  return cat16(lo, hi);
}

// Async-copy one 16x512 f16 core1 tile slice (this wave's 1/8) into LDS.
// Each of the 4 instructions: 32 lanes x 16B = 512B, globally contiguous.
// Per-lane LDS address re-inserts the +8-half row padding.
static __device__ __forceinline__ void async_tile(const _Float16* __restrict__ src_tile,
                                                  unsigned lds_base_bytes,
                                                  int w, int lane) {
  #pragma unroll
  for (int q = 0; q < 4; ++q) {
    unsigned u   = (unsigned)(w * 128 + q * 32 + lane);  // 8-half unit index
    unsigned i2  = u >> 6;                               // (u*8) / 512
    unsigned ko  = (u & 63u) * 8u;                       // (u*8) % 512
    unsigned lds = lds_base_bytes + (i2 * KP2 + ko) * 2u;
    unsigned gof = u * 16u;                              // bytes from tile start
    asm volatile("global_load_async_to_lds_b128 %0, %1, %2 offset:0"
                 :: "v"(lds), "v"(gof), "s"(src_tile)
                 : "memory");
  }
}

#define WAIT_ASYNC(n) asm volatile("s_wait_asynccnt " #n ::: "memory")

// ---- pre-pass: core1 f32 -> f16, re-laid out as W[t=r2][i2][k=r1*16+j2] ----
extern "C" __global__ __launch_bounds__(256)
void cvt_core1(const float* __restrict__ c1, _Float16* __restrict__ wdst) {
  int idx = blockIdx.x * 256 + threadIdx.x;   // 0..262143
  int t   = idx >> 13;
  int rem = idx & 8191;
  int i2  = rem >> 9;
  int k   = rem & 511;
  int r1  = k >> 4, j2 = k & 15;
  wdst[idx] = (_Float16)c1[r1 * 8192 + i2 * 512 + j2 * 32 + t];
}

extern "C" __global__ __launch_bounds__(256, 1)
void mpo_tt_kernel(const float* __restrict__ x,
                   const float* __restrict__ core0,
                   const float* __restrict__ core2,
                   const float* __restrict__ bias,
                   const _Float16* __restrict__ wsC1,
                   float* __restrict__ y)
{
  extern __shared__ _Float16 lds[];
  _Float16* XA  = lds + OFF_XA;
  _Float16* B0L = lds + OFF_B0;
  _Float16* B2L = lds + OFF_B2;
  _Float16* T1A = lds + OFF_T1;
  _Float16* SCR = lds + OFF_SC;

  const int tid  = threadIdx.x;
  const int lane = tid & 31;
  const int w    = tid >> 5;          // wave id == local i1
  const int b    = blockIdx.x;        // batch element
  const int half = blockIdx.y;        // i1 half (0/1)
  const int i1   = half * 8 + w;
  const int hh   = lane >> 4;         // half-wave (K split)
  const int nn   = lane & 15;         // M/N row/col within tile

  const unsigned c1b0 = (unsigned)(size_t)(lds + OFF_C1);

  // prologue: async-stage tiles 0 and 1 immediately (2 tiles / 8 ops in flight)
  async_tile(wsC1,        c1b0,               w, lane);
  async_tile(wsC1 + 8192, c1b0 + C1BUF_BYTES, w, lane);

  // ---- cooperative staging: XA, B0L, B2L (f32 -> f16, zero K pads) ----
  {
    const float* xb = x + (size_t)b * 4096;
    _Float16* dst = XA + tid * KP;    // row = j2*16 + j3 = tid
    #pragma unroll
    for (int j1 = 0; j1 < 16; ++j1) dst[j1] = (_Float16)xb[j1 * 256 + tid];
    #pragma unroll
    for (int k = 16; k < KP; ++k) dst[k] = (_Float16)0.0f;
  }
  {
    int i1s = tid >> 5, r1 = tid & 31;
    int i1g = half * 8 + i1s;
    _Float16* dst = B0L + (i1s * 32 + r1) * KP;
    #pragma unroll
    for (int j1 = 0; j1 < 16; ++j1)
      dst[j1] = (_Float16)core0[(i1g * 16 + j1) * 32 + r1];
    #pragma unroll
    for (int k = 16; k < KP; ++k) dst[k] = (_Float16)0.0f;
  }
  for (int rr = tid; rr < 512; rr += 256) {   // rr = r2*16 + i3
    _Float16* dst = B2L + rr * KP;
    #pragma unroll
    for (int j3 = 0; j3 < 16; ++j3) dst[j3] = (_Float16)core2[rr * 16 + j3];
    #pragma unroll
    for (int k = 16; k < KP; ++k) dst[k] = (_Float16)0.0f;
  }
  for (int e = tid; e < 8 * 16 * KP; e += 256) SCR[e] = (_Float16)0.0f;

  __syncthreads();

  // ---- stage 1: T1[j3][r1*16+j2] = sum_j1 x[b,j1,j2,j3] * core0[i1,j1,r1] ----
  // GEMM: M=(j2 j3) (j2 = m-tile, j3 = in-tile row), N=r1 (2 tiles), K=16 pad 32
  {
    _Float16* t1 = T1A + w * 16 * KP2;
    const _Float16* b0 = B0L + w * 32 * KP;
    #pragma unroll
    for (int nt = 0; nt < 2; ++nt) {
      v16h bf = frag(b0 + (nt * 16 + nn) * KP + hh * 8);
      for (int mt = 0; mt < 16; ++mt) {         // mt = j2
        v16h af = frag(XA + (mt * 16 + nn) * KP + hh * 8);
        v8f d = {};
        d = __builtin_amdgcn_wmma_f32_16x16x32_f16(false, af, false, bf,
                                                   (short)0, d, false, false);
        int kidx = (nt * 16 + nn) * 16 + mt;    // k = r1*16 + j2
        #pragma unroll
        for (int v = 0; v < 8; ++v) {
          int j3 = hh * 8 + v;                  // D: m = v + 8*hh
          t1[j3 * KP2 + kidx] = (_Float16)d[v];
        }
      }
    }
  }
  // no barrier needed: T1A slice is produced and consumed by this wave only

  // ---- hoist all 16 A-fragments (invariant across the whole t-loop) ----
  const _Float16* t1row = T1A + (w * 16 + nn) * KP2;   // A row m = j3 = nn
  v16h afr[16];
  #pragma unroll
  for (int kt = 0; kt < 16; ++kt)
    afr[kt] = frag(t1row + kt * 32 + hh * 8);

  // ---- stage 2 + fused stage 3; quad-buffered tiles, depth-2 prefetch,
  //      single barrier per iteration (reuse safety: buf[(t+2)&3] was last
  //      read at compute(t-2), which every wave finished before signaling
  //      barrier(t-1), which this wave passed before issuing) ----
  v8f yacc = {};
  _Float16* scr = SCR + w * 16 * KP;

  for (int t = 0; t < 32; ++t) {                       // t = r2 (n-tile)
    if (t + 2 < 32)
      async_tile(wsC1 + (size_t)(t + 2) * 8192,
                 c1b0 + (unsigned)((t + 2) & 3) * C1BUF_BYTES, w, lane);
    if      (t + 2 < 32) WAIT_ASYNC(0x8);  // tiles t+1,t+2 may be in flight
    else if (t + 1 < 32) WAIT_ASYNC(0x4);  // only tile t+1 in flight
    else                 WAIT_ASYNC(0x0);
    __syncthreads();

    // stage 2: acc[j3][i2] = sum_{r1,j2} T1 * core1   (K = 512 -> 16 k-steps)
    v8f acc = {};
    const _Float16* c1row = lds + OFF_C1 + (unsigned)(t & 3) * C1BUF_HALFS
                                + nn * KP2;
    #pragma unroll
    for (int kt = 0; kt < 16; ++kt) {
      v16h bf = frag(c1row + kt * 32 + hh * 8);
      acc = __builtin_amdgcn_wmma_f32_16x16x32_f16(false, afr[kt], false, bf,
                                                   (short)0, acc, false, false);
    }

    // fused stage 3: transpose D (j3 x i2) -> SCR as A (m=i2, k=j3), f16 packed
    #pragma unroll
    for (int v = 0; v < 8; v += 2) {
      h2 p;
      p.x = (_Float16)acc[v];
      p.y = (_Float16)acc[v + 1];
      *(h2*)(scr + nn * KP + hh * 8 + v) = p;          // scr[i2=nn][j3=hh*8+v]
    }
    v16h a3 = frag(scr + nn * KP + hh * 8);
    v16h b3 = frag(B2L + (t * 16 + nn) * KP + hh * 8); // B[j3][i3] for r2=t
    yacc = __builtin_amdgcn_wmma_f32_16x16x32_f16(false, a3, false, b3,
                                                  (short)0, yacc, false, false);
  }

  // ---- write y[b, i1, i2, i3] (+bias); D: m = i2, n = i3 ----
  {
    float* yb = y + (size_t)b * 4096 + i1 * 256;
    const float* bb = bias + i1 * 256;
    #pragma unroll
    for (int v = 0; v < 8; ++v) {
      int i2 = hh * 8 + v;
      int idx = i2 * 16 + nn;
      yb[idx] = yacc[v] + bb[idx];
    }
  }
}

extern "C" void kernel_launch(void* const* d_in, const int* in_sizes, int n_in,
                              void* d_out, int out_size, void* d_ws, size_t ws_size,
                              hipStream_t stream) {
  (void)n_in; (void)out_size; (void)ws_size;
  const float* x  = (const float*)d_in[0];
  const float* c0 = (const float*)d_in[1];
  const float* c1 = (const float*)d_in[2];
  const float* c2 = (const float*)d_in[3];
  const float* bs = (const float*)d_in[4];
  float* y = (float*)d_out;
  _Float16* wsC1 = (_Float16*)d_ws;           // 262144 halfs = 512KB scratch

  int B = in_sizes[0] / 4096;

  cvt_core1<<<1024, 256, 0, stream>>>(c1, wsC1);

  hipFuncSetAttribute((const void*)mpo_tt_kernel,
                      hipFuncAttributeMaxDynamicSharedMemorySize, LDS_BYTES);
  dim3 grid(B, 2);
  mpo_tt_kernel<<<grid, 256, LDS_BYTES, stream>>>(x, c0, c2, bs, wsC1, y);
}
